// CCCrossLayerAttentionV_76227079569757
// MI455X (gfx1250) — compile-verified, matrix operand order checked by reading, not run on hardware
//
#include <hip/hip_runtime.h>
#include <hip/hip_bf16.h>

typedef __attribute__((ext_vector_type(16))) _Float16 v16h;
typedef __attribute__((ext_vector_type(8)))  _Float16 v8h;
typedef __attribute__((ext_vector_type(8)))  float    v8f;

#define WD   128
#define HD   128
#define NPIX 16384   // pixels per batch image (128*128)
#define NEG_INF_F (-1000000000.0f)

// ---------------------------------------------------------------------------
// f32 -> f16 conversion (weights)
// ---------------------------------------------------------------------------
__global__ void f32_to_f16_kernel(const float* __restrict__ in,
                                  _Float16* __restrict__ out, int n) {
    int i = blockIdx.x * blockDim.x + threadIdx.x;
    if (i < n) out[i] = (_Float16)in[i];
}

// ---------------------------------------------------------------------------
// Bilinear upsample high (64x64 -> 128x128, half-pixel centers, edge clamp,
// matching jax.image.resize method='linear' on upscale) + concat low.
// Xcat layout: [b][pixel][channel0..511] f16; ch 0..255 = high_up, 256..511 = low.
// idx decomposed channel-fastest so the f16 writes are fully coalesced.
// ---------------------------------------------------------------------------
__global__ void upsample_concat_kernel(const float* __restrict__ high,
                                       const float* __restrict__ low,
                                       _Float16* __restrict__ Xcat) {
    long idx = (long)blockIdx.x * blockDim.x + threadIdx.x; // over 4*16384*512
    const long total = 4L * NPIX * 512;
    if (idx >= total) return;
    int c = (int)(idx & 511);
    int p = (int)((idx >> 9) & (NPIX - 1));
    int b = (int)(idx >> 23);
    float r;
    if (c < 256) {
        int w = p & 127, h = p >> 7;
        float sy = 0.5f * h - 0.25f;
        float sx = 0.5f * w - 0.25f;
        float fy0 = floorf(sy), fx0 = floorf(sx);
        float wy = sy - fy0, wx = sx - fx0;
        int y0 = (int)fy0, x0 = (int)fx0;
        int y0c = min(63, max(0, y0)),     x0c = min(63, max(0, x0));
        int y1c = min(63, max(0, y0 + 1)), x1c = min(63, max(0, x0 + 1));
        const float* src = high + ((long)b * 256 + c) * 4096;
        float v00 = src[y0c * 64 + x0c], v01 = src[y0c * 64 + x1c];
        float v10 = src[y1c * 64 + x0c], v11 = src[y1c * 64 + x1c];
        r = (1.f - wy) * ((1.f - wx) * v00 + wx * v01)
          +        wy  * ((1.f - wx) * v10 + wx * v11);
    } else {
        r = low[((long)b * 256 + (c - 256)) * NPIX + p];
    }
    Xcat[idx] = (_Float16)r;   // ((b*NPIX+p)*512 + c) == idx
}

// ---------------------------------------------------------------------------
// Operand assembly: per the ISA 16-bit 16x32 operand table, one lane's 16
// halfs for a 32-wide K step are two contiguous 16-byte chunks.
// ---------------------------------------------------------------------------
__device__ __forceinline__ v16h load_2x8h(const _Float16* p0, const _Float16* p1) {
    v8h lo = *(const v8h*)p0;
    v8h hi = *(const v8h*)p1;
    v16h r;
#pragma unroll
    for (int i = 0; i < 8; ++i) { r[i] = lo[i]; r[8 + i] = hi[i]; }
    return r;
}

// ---------------------------------------------------------------------------
// WMMA GEMM: D[b, n, m] = sum_k W[m,k] * X[b, n, k] (+bias).
//   W  : f16 row-major (M x K)
//   X  : f16 pixel-major [b][n][k]; rows k<KSPLIT from X0 (ld0), else X1 (ld1)
//   out: NK layout (f32 and/or f16) unless BN, then NCHW f32 + BN + ReLU.
// One wave per 16(M) x 64(N) tile: A loaded once per K-step, 4 wmma issue.
// K templated -> fully unrolled inner loop.
// ---------------------------------------------------------------------------
template<int K, int KSPLIT, bool BIAS, bool BN, bool O32, bool O16>
__global__ __launch_bounds__(32)
void wmma_gemm_kernel(const _Float16* __restrict__ Wt, int M,
                      const _Float16* __restrict__ X0, int ld0,
                      const _Float16* __restrict__ X1, int ld1,
                      const float* __restrict__ bias,
                      float* __restrict__ out32, _Float16* __restrict__ out16,
                      const float* __restrict__ bn_g, const float* __restrict__ bn_b,
                      const float* __restrict__ bn_m, const float* __restrict__ bn_v) {
    constexpr int NT = 4;                 // N tiles per wave
    const int N = NPIX;
    int n0   = blockIdx.x * (16 * NT);
    int m0   = blockIdx.y << 4;
    int b    = blockIdx.z;
    int lane = threadIdx.x;               // wave32
    int ml   = lane & 15;
    int khi  = (lane & 16) ? 8 : 0;       // lane-group K offset (ISA operand table)
    long bpix = (long)b * N;

    const _Float16* wrow = Wt + (long)(m0 + ml) * K;
    __builtin_prefetch(wrow, 0, 3);

    v8f acc[NT];
#pragma unroll
    for (int t = 0; t < NT; ++t) acc[t] = v8f{0.f,0.f,0.f,0.f,0.f,0.f,0.f,0.f};

#pragma unroll
    for (int kk = 0; kk < K; kk += 32) {
        const bool use1 = (KSPLIT < K) && (kk >= KSPLIT);   // compile-time foldable side
        v16h a = load_2x8h(wrow + kk + khi, wrow + kk + 16 + khi);
#pragma unroll
        for (int t = 0; t < NT; ++t) {
            int nn = n0 + t * 16 + ml;
            const _Float16* xr = use1
                ? (X1 + (bpix + nn) * (long)ld1 + (kk - KSPLIT))
                : (X0 + (bpix + nn) * (long)ld0 + kk);
            v16h bm = load_2x8h(xr + khi, xr + 16 + khi);
            acc[t] = __builtin_amdgcn_wmma_f32_16x16x32_f16(
                false, a, false, bm, (short)0, acc[t], false, false);
        }
    }

    // C/D layout: VGPR r -> M = m0 + 8*(lane>=16) + r, N = lane&15
    int mbase = m0 + ((lane >> 4) << 3);
#pragma unroll
    for (int t = 0; t < NT; ++t) {
        int nn = n0 + t * 16 + ml;
        long pix = bpix + nn;
        if constexpr (BN) {
#pragma unroll
            for (int r = 0; r < 8; ++r) {
                int mg = mbase + r;
                float val = acc[t][r];
                if constexpr (BIAS) val += bias[mg];
                float inv = rsqrtf(bn_v[mg] + 1e-5f);
                val = (val - bn_m[mg]) * (inv * bn_g[mg]) + bn_b[mg];
                val = fmaxf(val, 0.f);
                out32[((long)b * M + mg) * N + nn] = val;   // NCHW for d_out
            }
        } else {
            float vals[8];
#pragma unroll
            for (int r = 0; r < 8; ++r) {
                vals[r] = acc[t][r];
                if constexpr (BIAS) vals[r] += bias[mbase + r];
            }
            if constexpr (O32) {                 // 8 consecutive f32 -> 2x b128
                float4* dst = (float4*)(out32 + pix * M + mbase);
                dst[0] = make_float4(vals[0], vals[1], vals[2], vals[3]);
                dst[1] = make_float4(vals[4], vals[5], vals[6], vals[7]);
            }
            if constexpr (O16) {                 // 8 consecutive f16 -> 1x b128
                v8h h;
#pragma unroll
                for (int r = 0; r < 8; ++r) h[r] = (_Float16)vals[r];
                *(v8h*)(out16 + pix * M + mbase) = h;
            }
        }
    }
}

// ---------------------------------------------------------------------------
// Criss-cross attention, one block per (b,h,w), 128 threads. NK layouts:
//   pq/pk: [b][pixel][16] f32, pv/value: [b][pixel][128] f32 (value16 mirror).
// ---------------------------------------------------------------------------
__global__ __launch_bounds__(128)
void cc_attention_kernel(const float* __restrict__ pq, const float* __restrict__ pk,
                         const float* __restrict__ pv,
                         float* __restrict__ value32, _Float16* __restrict__ value16,
                         const float* __restrict__ gammaPtr) {
    __shared__ float pq_s[16];
    __shared__ float att[256];
    __shared__ float red[128];
    int w = blockIdx.x, h = blockIdx.y, b = blockIdx.z;
    int t = threadIdx.x;
    long bpix = (long)b * NPIX;

    if (t < 16) pq_s[t] = pq[(bpix + h * WD + w) * 16 + t];
    __syncthreads();

    const float* pkH = pk + (bpix + t * WD + w) * 16;   // pixel (x=t, w)
    const float* pkW = pk + (bpix + h * WD + t) * 16;   // pixel (h, x=t)
    float eH = 0.f, eW = 0.f;
#pragma unroll
    for (int c = 0; c < 16; ++c) {
        float q = pq_s[c];
        eH += q * pkH[c];
        eW += q * pkW[c];
    }
    if (t == h) eH += NEG_INF_F;

    red[t] = fmaxf(eH, eW);
    __syncthreads();
    for (int s = 64; s > 0; s >>= 1) {
        if (t < s) red[t] = fmaxf(red[t], red[t + s]);
        __syncthreads();
    }
    float mx = red[0];
    __syncthreads();

    float e0 = expf(eH - mx), e1 = expf(eW - mx);
    att[t] = e0; att[128 + t] = e1;
    red[t] = e0 + e1;
    __syncthreads();
    for (int s = 64; s > 0; s >>= 1) {
        if (t < s) red[t] += red[t + s];
        __syncthreads();
    }
    float inv = 1.f / red[0];
    att[t] *= inv; att[128 + t] *= inv;
    __syncthreads();

    // apply: thread t == channel c; per x-step the 128 threads read 128
    // contiguous floats (NK layout) -> fully coalesced.
    float accH = 0.f, accW = 0.f;
    for (int x = 0; x < 128; ++x) {
        accH += pv[(bpix + x * WD + w) * 128 + t] * att[x];
        accW += pv[(bpix + h * WD + x) * 128 + t] * att[128 + x];
    }
    float g = gammaPtr[0];
    long vi = (bpix + h * WD + w) * 128 + t;
    float nv = g * (accH + accW) + value32[vi];
    value32[vi] = nv;
    value16[vi] = (_Float16)nv;
}

// ---------------------------------------------------------------------------
extern "C" void kernel_launch(void* const* d_in, const int* in_sizes, int n_in,
                              void* d_out, int out_size, void* d_ws, size_t ws_size,
                              hipStream_t stream) {
    const float* low   = (const float*)d_in[0];
    const float* high  = (const float*)d_in[1];
    const float* Wc1   = (const float*)d_in[2];
    const float* bc1   = (const float*)d_in[3];
    const float* Wc2   = (const float*)d_in[4];
    const float* bc2   = (const float*)d_in[5];
    const float* Wq    = (const float*)d_in[6];
    const float* bq    = (const float*)d_in[7];
    const float* Wk    = (const float*)d_in[8];
    const float* bk    = (const float*)d_in[9];
    const float* Wv    = (const float*)d_in[10];
    const float* bv    = (const float*)d_in[11];
    const float* gamma = (const float*)d_in[12];
    const float* Wb    = (const float*)d_in[13];
    const float* bn_g  = (const float*)d_in[14];
    const float* bn_b  = (const float*)d_in[15];
    const float* bn_m  = (const float*)d_in[16];
    const float* bn_v  = (const float*)d_in[17];
    float* out = (float*)d_out;

    // workspace layout (256B-aligned slabs)
    char* base = (char*)d_ws;
    size_t off = 0;
    auto take = [&](size_t bytes) -> char* {
        char* p = base + off;
        off = (off + bytes + 255) & ~(size_t)255;
        return p;
    };
    _Float16* Xcat    = (_Float16*)take(4L * NPIX * 512 * 2);  // [b][pix][512]
    _Float16* query_h = (_Float16*)take(4L * NPIX * 128 * 2);  // [b][pix][128]
    float*    value32 = (float*)   take(4L * NPIX * 128 * 4);
    _Float16* value16 = (_Float16*)take(4L * NPIX * 128 * 2);
    float*    pq_f    = (float*)   take(4L * NPIX * 16  * 4);
    float*    pk_f    = (float*)   take(4L * NPIX * 16  * 4);
    float*    pv_f    = (float*)   take(4L * NPIX * 128 * 4);
    _Float16* Wc1h    = (_Float16*)take(128L * 512 * 2);
    _Float16* Wc2h    = (_Float16*)take(128L * 256 * 2);
    _Float16* Wqh     = (_Float16*)take(16L  * 128 * 2);
    _Float16* Wkh     = (_Float16*)take(16L  * 128 * 2);
    _Float16* Wvh     = (_Float16*)take(128L * 128 * 2);
    _Float16* Wbh     = (_Float16*)take(256L * 384 * 2);
    (void)ws_size; (void)in_sizes; (void)n_in; (void)out_size;

    auto cvt = [&](const float* src, _Float16* dst, int n) {
        f32_to_f16_kernel<<<(n + 255) / 256, 256, 0, stream>>>(src, dst, n);
    };
    cvt(Wc1, Wc1h, 128 * 512);
    cvt(Wc2, Wc2h, 128 * 256);
    cvt(Wq,  Wqh,  16 * 128);
    cvt(Wk,  Wkh,  16 * 128);
    cvt(Wv,  Wvh,  128 * 128);
    cvt(Wb,  Wbh,  256 * 384);

    {
        long total = 4L * NPIX * 512;
        upsample_concat_kernel<<<(unsigned)((total + 255) / 256), 256, 0, stream>>>(
            high, low, Xcat);
    }

    dim3 blk(32);
    const int GX = NPIX / 64;   // 4 N-tiles per wave

    // query = Wc1(128x512) * Xcat -> f16 NK
    wmma_gemm_kernel<512, 512, true, false, false, true>
        <<<dim3(GX, 8, 4), blk, 0, stream>>>(
            Wc1h, 128, Xcat, 512, Xcat, 512, bc1,
            nullptr, query_h, nullptr, nullptr, nullptr, nullptr);

    // value = Wc2(128x256) * high_up (Xcat ch 0..255) -> f32 + f16 NK
    wmma_gemm_kernel<256, 256, true, false, true, true>
        <<<dim3(GX, 8, 4), blk, 0, stream>>>(
            Wc2h, 128, Xcat, 512, Xcat, 512, bc2,
            value32, value16, nullptr, nullptr, nullptr, nullptr);

    // pq = Wq(16x128) * query (loop-invariant)
    wmma_gemm_kernel<128, 128, true, false, true, false>
        <<<dim3(GX, 1, 4), blk, 0, stream>>>(
            Wqh, 16, query_h, 128, query_h, 128, bq,
            pq_f, nullptr, nullptr, nullptr, nullptr, nullptr);

    for (int it = 0; it < 2; ++it) {
        wmma_gemm_kernel<128, 128, true, false, true, false>
            <<<dim3(GX, 1, 4), blk, 0, stream>>>(
                Wkh, 16, value16, 128, value16, 128, bk,
                pk_f, nullptr, nullptr, nullptr, nullptr, nullptr);
        wmma_gemm_kernel<128, 128, true, false, true, false>
            <<<dim3(GX, 8, 4), blk, 0, stream>>>(
                Wvh, 128, value16, 128, value16, 128, bv,
                pv_f, nullptr, nullptr, nullptr, nullptr, nullptr);
        cc_attention_kernel<<<dim3(WD, HD, 4), 128, 0, stream>>>(
            pq_f, pk_f, pv_f, value32, value16, gamma);
    }

    // fused = Wb(256x384) * concat(value, high_up); BN + ReLU -> d_out (NCHW)
    wmma_gemm_kernel<384, 128, false, true, true, false>
        <<<dim3(GX, 16, 4), blk, 0, stream>>>(
            Wbh, 256, value16, 128, Xcat, 512, nullptr,
            out, nullptr, bn_g, bn_b, bn_m, bn_v);
}
